// StackedBilstmCrfModel_24893630448026
// MI455X (gfx1250) — compile-verified, matrix operand order checked by reading, not run on hardware
//
#include <hip/hip_runtime.h>

// ---------------------------------------------------------------------------
// Stacked BiLSTM (B=64, T=512, E=300, H=512) for MI455X / gfx1250.
// bf16 WMMA (v_wmma_f32_16x16x32_bf16) with f32 accumulation.
// ---------------------------------------------------------------------------

#define BATCH 64
#define TLEN  512
#define EDIM  300
#define HDIM  512
#define NGATE 2048            // 4*H
#define KX0   300             // layer-0 x width
#define KX0P  320             // padded to multiple of 32
#define KTOT0 (KX0P + HDIM)   // 832
#define KX1   1024            // layer-1 x width (2H), already 32-aligned
#define KTOT1 (KX1 + HDIM)    // 1536

typedef __attribute__((ext_vector_type(16))) __bf16 v16bf;
typedef __attribute__((ext_vector_type(8)))  __bf16 v8bf;
typedef __attribute__((ext_vector_type(8)))  float  v8f;

// ---------------------------------------------------------------------------
// device helpers
// ---------------------------------------------------------------------------
__device__ __forceinline__ float sigmoidf_(float x) {
    return 1.0f / (1.0f + __expf(-x));
}
__device__ __forceinline__ float tanhf_(float x) {
    float xc = fminf(fmaxf(x, -15.0f), 15.0f);
    float e  = __expf(2.0f * xc);
    return (e - 1.0f) / (e + 1.0f);
}

// Load a 16-element bf16 fragment for the 16x16x32 WMMA.
// Per ISA layout: lanes 0-15 hold K = k0+{0..7, 16..23}, lanes 16-31 hold
// K = k0+{8..15, 24..31}; both map to the same row (lane & 15).
// rowPtr points at the start of that lane's row; rows are 16B aligned.
__device__ __forceinline__ v16bf load_frag(const __bf16* rowPtr, int k0, int kh) {
    v8bf lo = *(const v8bf*)(rowPtr + k0 + kh * 8);
    v8bf hi = *(const v8bf*)(rowPtr + k0 + 16 + kh * 8);
    v16bf r;
#pragma unroll
    for (int i = 0; i < 8; ++i) { r[i] = lo[i]; r[i + 8] = hi[i]; }
    return r;
}

// ---------------------------------------------------------------------------
// One recurrent LSTM step for one direction:
//   z = [x_t , h_prev] @ W + b ; gate math ; length masking ; (reverse) scatter
// 128 waves: wave = mTile (4 batch tiles) x nTile (32 hidden-unit tiles).
// Each wave computes four 16x16 gate tiles (i,j,f,o) sharing one A fragment.
// Weights are pre-transposed bf16: Wt[n][k], row stride = Ktot (16B aligned).
// ---------------------------------------------------------------------------
template <typename OUT_T>
__global__ __launch_bounds__(256)
void lstm_step_kernel(const __bf16* __restrict__ x, int xStride,
                      const __bf16* __restrict__ Wt, int Ktot,
                      const float*  __restrict__ bias,
                      const __bf16* __restrict__ hPrev,
                      __bf16*       __restrict__ hNext,
                      float*        __restrict__ cState,
                      OUT_T*        __restrict__ out, int outStride, int outOff,
                      const int*    __restrict__ lens,
                      int s, int rev)
{
    const int tid   = threadIdx.x;
    const int lane  = tid & 31;
    const int wave  = blockIdx.x * (blockDim.x >> 5) + (tid >> 5);  // 0..127
    const int mTile = wave & 3;     // batch tile (16 rows each)
    const int nTile = wave >> 2;    // hidden-unit tile (16 cols each), 0..31
    const int kh    = lane >> 4;    // K half select
    const int r     = lane & 15;

    const int bRow = mTile * 16 + r;       // batch row this lane loads for A
    const int nCol = nTile * 16 + r;       // hidden column for B frags & C/D

    // per-batch (reversed) time index for the gather on x
    const int Lrow = lens[bRow];
    const int tIdx = rev ? ((s < Lrow) ? (Lrow - 1 - s) : s) : s;
    const __bf16* xRow = x + ((size_t)bRow * TLEN + tIdx) * (size_t)xStride;
    const __bf16* hRow = hPrev + (size_t)bRow * HDIM;

    const __bf16* wRow[4];
#pragma unroll
    for (int g = 0; g < 4; ++g)
        wRow[g] = Wt + (size_t)(g * HDIM + nCol) * (size_t)Ktot;

    // accumulators init = bias (every row of a column tile gets bias[g*H+n])
    v8f acc[4];
#pragma unroll
    for (int g = 0; g < 4; ++g) {
        float bv = bias[g * HDIM + nCol];
#pragma unroll
        for (int i = 0; i < 8; ++i) acc[g][i] = bv;
    }

    const int Kx = Ktot - HDIM;

    // x-part of the concatenated GEMM
    for (int k0 = 0; k0 < Kx; k0 += 32) {
        v16bf a = load_frag(xRow, k0, kh);
#pragma unroll
        for (int g = 0; g < 4; ++g) {
            v16bf b = load_frag(wRow[g], k0, kh);
            acc[g] = __builtin_amdgcn_wmma_f32_16x16x32_bf16(
                false, a, false, b, (short)0, acc[g], false, false);
        }
    }
    // h-part (recurrent)
    for (int k0 = 0; k0 < HDIM; k0 += 32) {
        v16bf a = load_frag(hRow, k0, kh);
#pragma unroll
        for (int g = 0; g < 4; ++g) {
            v16bf b = load_frag(wRow[g], Kx + k0, kh);
            acc[g] = __builtin_amdgcn_wmma_f32_16x16x32_bf16(
                false, a, false, b, (short)0, acc[g], false, false);
        }
    }

    // fused LSTM cell. C/D layout: VGPR i -> row mTile*16 + kh*8 + i, col nCol.
#pragma unroll
    for (int i = 0; i < 8; ++i) {
        const int  bb    = mTile * 16 + kh * 8 + i;
        const int  Lb    = lens[bb];
        const bool valid = (s < Lb);
        const size_t idx = (size_t)bb * HDIM + nCol;

        float ig = acc[0][i], jg = acc[1][i], fg = acc[2][i], og = acc[3][i];
        float cPrev  = cState[idx];
        float hPrevV = (float)hPrev[idx];
        float cNew   = sigmoidf_(fg + 1.0f) * cPrev + sigmoidf_(ig) * tanhf_(jg);
        float hNewV  = sigmoidf_(og) * tanhf_(cNew);

        cState[idx] = valid ? cNew : cPrev;
        hNext[idx]  = (__bf16)(valid ? hNewV : hPrevV);

        const int tOut = rev ? ((s < Lb) ? (Lb - 1 - s) : s) : s;
        out[((size_t)bb * TLEN + tOut) * (size_t)outStride + outOff + nCol] =
            (OUT_T)(valid ? hNewV : 0.0f);
    }
}

// ---------------------------------------------------------------------------
// conversion / init kernels
// ---------------------------------------------------------------------------
__global__ void cvt_emb_kernel(const float* __restrict__ emb,
                               __bf16* __restrict__ dst)
{
    size_t i = (size_t)blockIdx.x * blockDim.x + threadIdx.x;
    size_t total = (size_t)BATCH * TLEN * KX0P;
    if (i >= total) return;
    int k = (int)(i % KX0P);
    size_t bt = i / KX0P;
    float v = (k < EDIM) ? emb[bt * EDIM + k] : 0.0f;
    dst[i] = (__bf16)v;
}

// W is f32 [(Kx + H) x 4H] row-major; produce bf16 Wt [4H x Ktot] with the
// x-rows zero-padded from Kx to Kxp (Ktot = Kxp + H), transposed so B-fragment
// loads along K are contiguous.
__global__ void cvt_weight_kernel(const float* __restrict__ W,
                                  __bf16* __restrict__ Wt,
                                  int Kx, int Kxp, int Ktot)
{
    size_t i = (size_t)blockIdx.x * blockDim.x + threadIdx.x;
    size_t total = (size_t)NGATE * (size_t)Ktot;
    if (i >= total) return;
    int kout = (int)(i % Ktot);
    int n    = (int)(i / Ktot);
    int kin;
    if (kout < Kxp) kin = (kout < Kx) ? kout : -1;
    else            kin = Kx + (kout - Kxp);
    float v = (kin >= 0) ? W[(size_t)kin * NGATE + n] : 0.0f;
    Wt[i] = (__bf16)v;
}

__global__ void zero_state_kernel(float* __restrict__ c,
                                  __bf16* __restrict__ hA,
                                  __bf16* __restrict__ hB)
{
    int i = blockIdx.x * blockDim.x + threadIdx.x;
    if (i < BATCH * HDIM) {
        c[i]  = 0.0f;
        hA[i] = (__bf16)0.0f;
        hB[i] = (__bf16)0.0f;
    }
}

// ---------------------------------------------------------------------------
// host-side scan driver (all launches on `stream`; graph-capture safe)
// ---------------------------------------------------------------------------
template <typename OUT_T>
static void run_scan(const __bf16* x, int xStride,
                     const __bf16* Wt, int Ktot,
                     const float* bias,
                     float* cS, __bf16* hA, __bf16* hB,
                     OUT_T* out, int outStride, int outOff,
                     const int* lens, int rev, hipStream_t stream)
{
    zero_state_kernel<<<(BATCH * HDIM + 255) / 256, 256, 0, stream>>>(cS, hA, hB);
    for (int s = 0; s < TLEN; ++s) {
        const __bf16* hp = (s & 1) ? hB : hA;
        __bf16*       hn = (s & 1) ? hA : hB;
        lstm_step_kernel<OUT_T><<<16, 256, 0, stream>>>(
            x, xStride, Wt, Ktot, bias, hp, hn, cS,
            out, outStride, outOff, lens, s, rev);
    }
}

extern "C" void kernel_launch(void* const* d_in, const int* in_sizes, int n_in,
                              void* d_out, int out_size, void* d_ws, size_t ws_size,
                              hipStream_t stream)
{
    const float* emb  = (const float*)d_in[0];
    const int*   lens = (const int*)  d_in[1];
    const float* Wf0  = (const float*)d_in[2];
    const float* bf0  = (const float*)d_in[3];
    const float* Wb0  = (const float*)d_in[4];
    const float* bb0  = (const float*)d_in[5];
    const float* Wf1  = (const float*)d_in[6];
    const float* bf1  = (const float*)d_in[7];
    const float* Wb1  = (const float*)d_in[8];
    const float* bb1  = (const float*)d_in[9];
    float* out = (float*)d_out;

    // carve workspace (256B aligned slices)
    char* p = (char*)d_ws;
    auto carve = [&](size_t bytes) -> char* {
        char* q = p;
        p += (bytes + 255) & ~(size_t)255;
        return q;
    };
    __bf16* embPad = (__bf16*)carve((size_t)BATCH * TLEN * KX0P * 2);      // 20 MB
    __bf16* out0   = (__bf16*)carve((size_t)BATCH * TLEN * 2 * HDIM * 2);  // 64 MB
    __bf16* WtF0   = (__bf16*)carve((size_t)NGATE * KTOT0 * 2);
    __bf16* WtB0   = (__bf16*)carve((size_t)NGATE * KTOT0 * 2);
    __bf16* WtF1   = (__bf16*)carve((size_t)NGATE * KTOT1 * 2);
    __bf16* WtB1   = (__bf16*)carve((size_t)NGATE * KTOT1 * 2);
    float*  cS     = (float*) carve((size_t)BATCH * HDIM * 4);
    __bf16* hA     = (__bf16*)carve((size_t)BATCH * HDIM * 2);
    __bf16* hB     = (__bf16*)carve((size_t)BATCH * HDIM * 2);

    // one-time (per call) bf16 conversion / transposition
    {
        size_t n = (size_t)BATCH * TLEN * KX0P;
        cvt_emb_kernel<<<(int)((n + 255) / 256), 256, 0, stream>>>(emb, embPad);
    }
    {
        size_t n = (size_t)NGATE * KTOT0;
        int g = (int)((n + 255) / 256);
        cvt_weight_kernel<<<g, 256, 0, stream>>>(Wf0, WtF0, KX0, KX0P, KTOT0);
        cvt_weight_kernel<<<g, 256, 0, stream>>>(Wb0, WtB0, KX0, KX0P, KTOT0);
    }
    {
        size_t n = (size_t)NGATE * KTOT1;
        int g = (int)((n + 255) / 256);
        cvt_weight_kernel<<<g, 256, 0, stream>>>(Wf1, WtF1, KX1, KX1, KTOT1);
        cvt_weight_kernel<<<g, 256, 0, stream>>>(Wb1, WtB1, KX1, KX1, KTOT1);
    }

    // layer 0: out0[b,t, 0:H] = fw, [H:2H] = bw (bf16, feeds layer 1)
    run_scan<__bf16>(embPad, KX0P, WtF0, KTOT0, bf0, cS, hA, hB,
                     out0, 2 * HDIM, 0,    lens, 0, stream);
    run_scan<__bf16>(embPad, KX0P, WtB0, KTOT0, bb0, cS, hA, hB,
                     out0, 2 * HDIM, HDIM, lens, 1, stream);

    // layer 1: writes final f32 output [B, T, 2H]
    run_scan<float>(out0, 2 * HDIM, WtF1, KTOT1, bf1, cS, hA, hB,
                    out, 2 * HDIM, 0,    lens, 0, stream);
    run_scan<float>(out0, 2 * HDIM, WtB1, KTOT1, bb1, cS, hA, hB,
                    out, 2 * HDIM, HDIM, lens, 1, stream);

    (void)in_sizes; (void)n_in; (void)out_size; (void)ws_size;
}
